// EuclideanDeconf_12378095747956
// MI455X (gfx1250) — matrix operand order repaired and checked by reading, not compiled, not run on hardware
//
#include <hip/hip_runtime.h>

// Problem dims (fixed by the reference)
#define B_DIM 16384
#define D_DIM 1024
#define C_DIM 2048

// GEMM tiling
#define BM 128
#define BN 256
#define BK 32
#define LDS_STRIDE 80                     // bytes per K-row of 32 bf16 (64B) + 16B pad
#define A_BYTES (BM * LDS_STRIDE)         // 10240
#define B_BYTES (BN * LDS_STRIDE)         // 20480
#define STAGE_BYTES (A_BYTES + B_BYTES)   // 30720

// Workspace layout (bytes)
#define WS_X2_OFF 0u
#define WS_W2_OFF ((size_t)B_DIM * 4)
#define WS_XB_OFF (WS_W2_OFF + (size_t)C_DIM * 4)            // 73728
#define WS_WB_OFF (WS_XB_OFF + (size_t)B_DIM * D_DIM * 2)    // +32MB
#define WS_NEEDED (WS_WB_OFF + (size_t)C_DIM * D_DIM * 2)    // ~36.1MB

typedef __attribute__((ext_vector_type(16))) __bf16          v16bf;
typedef __attribute__((ext_vector_type(8)))  float           v8f;
typedef __attribute__((ext_vector_type(8)))  unsigned short  v8us;
typedef __attribute__((ext_vector_type(4)))  int             v4i;

// Address-space qualified 16B vector types for the async-DMA builtin.
typedef __attribute__((address_space(1))) v4i av4i_global;
typedef __attribute__((address_space(3))) v4i av4i_lds;

union FragB16 {
  v16bf v;
  v8us  u[2];
};

// fp32 pair -> packed bf16 (RNE) in one VALU op (verified to assemble on gfx1250).
static __device__ __forceinline__ unsigned pack2_bf16(float a, float b) {
  unsigned r;
  asm("v_cvt_pk_bf16_f32 %0, %1, %2" : "=v"(r) : "v"(a), "v"(b));
  return r;
}

// Async global->LDS 16B copy (ASYNCcnt-tracked DMA; no VGPR data path).
static __device__ __forceinline__ void async_copy_b128(const void* g, void* l) {
#if __has_builtin(__builtin_amdgcn_global_load_async_to_lds_b128)
  __builtin_amdgcn_global_load_async_to_lds_b128(
      (av4i_global*)(v4i*)g, (av4i_lds*)(v4i*)l, 0, 0);
#else
  *(uint4*)l = *(const uint4*)g;   // fallback: synchronous copy via VGPRs
#endif
}

static __device__ __forceinline__ void wait_async0() {
#if __has_builtin(__builtin_amdgcn_s_wait_asynccnt)
  __builtin_amdgcn_s_wait_asynccnt(0);
#else
  asm volatile("s_wait_asynccnt 0x0" ::: "memory");
#endif
}

// ---------------------------------------------------------------------------
// Kernel 1: per-row sums-of-squares (scaled by 1/D) for x and W, and
// (optionally) a bf16 transcode of x and W into workspace.
// One wave per row; 8 waves per block.
// ---------------------------------------------------------------------------
__global__ __launch_bounds__(256) void ed_prep(const float* __restrict__ x,
                                               const float* __restrict__ W,
                                               float* __restrict__ x2,
                                               float* __restrict__ w2,
                                               unsigned short* __restrict__ xb,
                                               unsigned short* __restrict__ wb,
                                               int writeBf16) {
  const int lane = threadIdx.x & 31;
  const int wave = threadIdx.x >> 5;
  const int row  = blockIdx.x * 8 + wave;          // 0 .. B_DIM + C_DIM - 1

  const float* src;
  unsigned short* dstb;
  float* dst;
  if (row < B_DIM) {
    src  = x + (size_t)row * D_DIM;
    dstb = xb + (size_t)row * D_DIM;
    dst  = x2 + row;
  } else {
    src  = W + (size_t)(row - B_DIM) * D_DIM;
    dstb = wb + (size_t)(row - B_DIM) * D_DIM;
    dst  = w2 + (row - B_DIM);
  }

  float s = 0.0f;
#pragma unroll
  for (int it = 0; it < D_DIM / 128; ++it) {       // 8 iterations of float4
    const int idx = it * 128 + lane * 4;
    const float4 v = *(const float4*)(src + idx);
    s += v.x * v.x + v.y * v.y + v.z * v.z + v.w * v.w;
    if (writeBf16) {
      uint2 p;
      p.x = pack2_bf16(v.x, v.y);
      p.y = pack2_bf16(v.z, v.w);
      *(uint2*)(dstb + idx) = p;
    }
  }
#pragma unroll
  for (int off = 16; off > 0; off >>= 1)
    s += __shfl_xor(s, off, 32);

  if (lane == 0) *dst = s * (1.0f / (float)D_DIM);
}

// ---------------------------------------------------------------------------
// Shared WMMA compute + epilogue (identical for both staging variants).
// ---------------------------------------------------------------------------
static __device__ __forceinline__ void wmma_step(const unsigned char* aB,
                                                 const unsigned char* bB,
                                                 int waveM, int waveN,
                                                 int lr, int lh,
                                                 v8f acc[4][4]) {
  FragB16 fa[4], fb[4];
  // A fragments (16x32 bf16): per lane, 8 halves at K-chunk, 8 at K-chunk+16
#pragma unroll
  for (int mi = 0; mi < 4; ++mi) {
    const unsigned char* p =
        aB + (waveM * 64 + mi * 16 + lr) * LDS_STRIDE + lh * 16;
    fa[mi].u[0] = *(const v8us*)p;
    fa[mi].u[1] = *(const v8us*)(p + 32);
  }
  // B fragments (32x16 bf16): per lane, 8 halves at K-chunk, 8 at K-chunk+16
#pragma unroll
  for (int ni = 0; ni < 4; ++ni) {
    const unsigned char* p =
        bB + (waveN * 64 + ni * 16 + lr) * LDS_STRIDE + lh * 16;
    fb[ni].u[0] = *(const v8us*)p;
    fb[ni].u[1] = *(const v8us*)(p + 32);
  }
#pragma unroll
  for (int mi = 0; mi < 4; ++mi)
#pragma unroll
    for (int ni = 0; ni < 4; ++ni)
      acc[mi][ni] = __builtin_amdgcn_wmma_f32_16x16x32_bf16(
          false, fa[mi].v, false, fb[ni].v, (short)0, acc[mi][ni],
          false, false);
}

static __device__ __forceinline__ void epilogue(v8f acc[4][4],
                                                const float* __restrict__ x2,
                                                const float* __restrict__ w2,
                                                float* __restrict__ out,
                                                int blockM, int blockN,
                                                int waveM, int waveN,
                                                int lr, int lh) {
  const float scale = 2.0f / (float)D_DIM;
#pragma unroll
  for (int mi = 0; mi < 4; ++mi) {
    const int mBase = blockM + waveM * 64 + mi * 16 + lh * 8;
#pragma unroll
    for (int ni = 0; ni < 4; ++ni) {
      const int n = blockN + waveN * 64 + ni * 16 + lr;
      const float wn = w2[n];
#pragma unroll
      for (int g = 0; g < 8; ++g) {
        const int m = mBase + g;
        out[(size_t)m * C_DIM + n] = acc[mi][ni][g] * scale - x2[m] - wn;
      }
    }
  }
}

// ---------------------------------------------------------------------------
// Kernel 2a: GEMM with bf16 sources, staged via async global->LDS DMA.
// Block tile 128x256, 8 waves (2x4), wave tile 64x64 (4x4 WMMA 16x16x32 bf16).
// ---------------------------------------------------------------------------
__global__ __launch_bounds__(256) void ed_gemm_bf16(
    const unsigned short* __restrict__ xb, const unsigned short* __restrict__ wb,
    const float* __restrict__ x2, const float* __restrict__ w2,
    float* __restrict__ out) {
  __shared__ __align__(16) unsigned char smem[2 * STAGE_BYTES];

  const int tid   = threadIdx.x;
  const int lane  = tid & 31;
  const int wave  = tid >> 5;
  const int waveM = wave & 1;
  const int waveN = wave >> 1;
  const int blockM = blockIdx.x * BM;
  const int blockN = blockIdx.y * BN;
  const int lr = lane & 15;
  const int lh = lane >> 4;

  // staging mapping: thread -> (row stripe of 64, 16B chunk along K)
  const int kq = tid & 3;        // 4 x 16B chunks per 64B K-row
  const int rr = tid >> 2;       // 0..63

  v8f acc[4][4];
#pragma unroll
  for (int mi = 0; mi < 4; ++mi)
#pragma unroll
    for (int ni = 0; ni < 4; ++ni)
#pragma unroll
      for (int g = 0; g < 8; ++g)
        acc[mi][ni][g] = 0.0f;

  // Issue DMA copies for K-step ks into LDS buffer `buf` (no VGPR data).
  auto astage = [&](int ks, int buf) {
    unsigned char* aB = smem + buf * STAGE_BYTES;
    unsigned char* bB = aB + A_BYTES;
    const unsigned short* ax =
        xb + (size_t)(blockM + rr) * D_DIM + ks * BK + kq * 8;
    unsigned char* al = aB + rr * LDS_STRIDE + kq * 16;
#pragma unroll
    for (int p = 0; p < 2; ++p)
      async_copy_b128(ax + (size_t)p * 64 * D_DIM, al + p * 64 * LDS_STRIDE);
    const unsigned short* bw =
        wb + (size_t)(blockN + rr) * D_DIM + ks * BK + kq * 8;
    unsigned char* bl = bB + rr * LDS_STRIDE + kq * 16;
#pragma unroll
    for (int p = 0; p < 4; ++p)
      async_copy_b128(bw + (size_t)p * 64 * D_DIM, bl + p * 64 * LDS_STRIDE);
  };

  const int KSTEPS = D_DIM / BK;   // 32
  astage(0, 0);
  for (int ks = 0; ks < KSTEPS; ++ks) {
    const int buf = ks & 1;
    wait_async0();                 // my stage(ks) writes to LDS are complete
    __syncthreads();               // everyone's stage(ks) visible; buf^1 free
    if (ks + 1 < KSTEPS) astage(ks + 1, buf ^ 1);   // DMA overlaps compute
    const unsigned char* aB = smem + buf * STAGE_BYTES;
    const unsigned char* bB = aB + A_BYTES;
    wmma_step(aB, bB, waveM, waveN, lr, lh, acc);
  }
  epilogue(acc, x2, w2, out, blockM, blockN, waveM, waveN, lr, lh);
}

// ---------------------------------------------------------------------------
// Kernel 2b: GEMM converting fp32 sources on the fly (small-workspace path).
// ---------------------------------------------------------------------------
__global__ __launch_bounds__(256) void ed_gemm_f32(
    const float* __restrict__ x, const float* __restrict__ W,
    const float* __restrict__ x2, const float* __restrict__ w2,
    float* __restrict__ out) {
  __shared__ __align__(16) unsigned char smem[2 * STAGE_BYTES];

  const int tid   = threadIdx.x;
  const int lane  = tid & 31;
  const int wave  = tid >> 5;
  const int waveM = wave & 1;
  const int waveN = wave >> 1;
  const int blockM = blockIdx.x * BM;
  const int blockN = blockIdx.y * BN;
  const int lr = lane & 15;
  const int lh = lane >> 4;

  const int kv = tid & 7;        // 8 float4 per 32-wide K row
  const int r0 = tid >> 3;       // 0..31 row stripe

  v8f acc[4][4];
#pragma unroll
  for (int mi = 0; mi < 4; ++mi)
#pragma unroll
    for (int ni = 0; ni < 4; ++ni)
#pragma unroll
      for (int g = 0; g < 8; ++g)
        acc[mi][ni][g] = 0.0f;

  uint2 aReg[4];
  uint2 bReg[8];

  auto gload = [&](int ks) {
    const float* ax = x + (size_t)(blockM + r0) * D_DIM + ks * BK + kv * 4;
#pragma unroll
    for (int p = 0; p < 4; ++p) {
      const float4 v = *(const float4*)(ax + (size_t)p * 32 * D_DIM);
      aReg[p].x = pack2_bf16(v.x, v.y);
      aReg[p].y = pack2_bf16(v.z, v.w);
    }
    const float* bw = W + (size_t)(blockN + r0) * D_DIM + ks * BK + kv * 4;
#pragma unroll
    for (int p = 0; p < 8; ++p) {
      const float4 v = *(const float4*)(bw + (size_t)p * 32 * D_DIM);
      bReg[p].x = pack2_bf16(v.x, v.y);
      bReg[p].y = pack2_bf16(v.z, v.w);
    }
  };

  auto lstore = [&](int buf) {
    unsigned char* aB = smem + buf * STAGE_BYTES;
    unsigned char* bB = aB + A_BYTES;
#pragma unroll
    for (int p = 0; p < 4; ++p)
      *(uint2*)(aB + (r0 + p * 32) * LDS_STRIDE + kv * 8) = aReg[p];
#pragma unroll
    for (int p = 0; p < 8; ++p)
      *(uint2*)(bB + (r0 + p * 32) * LDS_STRIDE + kv * 8) = bReg[p];
  };

  const int KSTEPS = D_DIM / BK;   // 32
  gload(0);
  for (int ks = 0; ks < KSTEPS; ++ks) {
    const int buf = ks & 1;
    lstore(buf);
    __syncthreads();
    if (ks + 1 < KSTEPS) gload(ks + 1);
    const unsigned char* aB = smem + buf * STAGE_BYTES;
    const unsigned char* bB = aB + A_BYTES;
    wmma_step(aB, bB, waveM, waveN, lr, lh, acc);
  }
  epilogue(acc, x2, w2, out, blockM, blockN, waveM, waveN, lr, lh);
}

// ---------------------------------------------------------------------------
extern "C" void kernel_launch(void* const* d_in, const int* in_sizes, int n_in,
                              void* d_out, int out_size, void* d_ws, size_t ws_size,
                              hipStream_t stream) {
  (void)in_sizes; (void)n_in; (void)out_size;
  const float* x = (const float*)d_in[0];   // [B, D] fp32
  const float* W = (const float*)d_in[1];   // [C, D] fp32
  float* out = (float*)d_out;               // [B, C] fp32

  unsigned char* ws = (unsigned char*)d_ws;
  float* x2 = (float*)(ws + WS_X2_OFF);                 // [B]
  float* w2 = (float*)(ws + WS_W2_OFF);                 // [C]
  unsigned short* xb = (unsigned short*)(ws + WS_XB_OFF);
  unsigned short* wb = (unsigned short*)(ws + WS_WB_OFF);

  const bool big = ws_size >= WS_NEEDED;

  ed_prep<<<(B_DIM + C_DIM) / 8, 256, 0, stream>>>(x, W, x2, w2, xb, wb,
                                                   big ? 1 : 0);

  dim3 grid(B_DIM / BM, C_DIM / BN);        // 128 x 8
  if (big)
    ed_gemm_bf16<<<grid, 256, 0, stream>>>(xb, wb, x2, w2, out);
  else
    ed_gemm_f32<<<grid, 256, 0, stream>>>(x, W, x2, w2, out);
}
